// SAGE_PYG_58110907515586
// MI455X (gfx1250) — matrix operand order repaired and checked by reading, not compiled, not run on hardware
//
#include <hip/hip_runtime.h>

// Problem constants (fixed by the reference).
#define NN      100000    // nodes
#define FIN     128       // in feats == hidden
#define NHID    128
#define NCLS    64
#define NE      1600000   // edges

typedef __attribute__((ext_vector_type(16))) __bf16 v16bf;
typedef __attribute__((ext_vector_type(8)))  float  v8f;

// ---------------------------------------------------------------- utilities
__global__ void zero_f32_kernel(float* __restrict__ p, size_t n) {
  size_t i = (size_t)blockIdx.x * blockDim.x + threadIdx.x;
  size_t stride = (size_t)gridDim.x * blockDim.x;
  for (; i < n; i += stride) p[i] = 0.0f;
}

__global__ void degree_kernel(const long long* __restrict__ dst,
                              float* __restrict__ deg, int nE) {
  int e = blockIdx.x * blockDim.x + threadIdx.x;
  if (e < nE) atomicAdd(&deg[(int)dst[e]], 1.0f);
}

// Wave-per-edge scatter-add: lane l adds 4 consecutive floats (D==128).
__global__ void scatter128_kernel(const float* __restrict__ x,
                                  const long long* __restrict__ src,
                                  const long long* __restrict__ dst,
                                  float* __restrict__ agg, int nE) {
  int e    = blockIdx.x * 8 + (threadIdx.x >> 5);   // 8 waves / 256-thread block
  int lane = threadIdx.x & 31;
  if (e >= nE) return;
  int s = (int)src[e], d = (int)dst[e];
  float4 v = ((const float4*)(x + (size_t)s * 128))[lane];
  float* a = agg + (size_t)d * 128 + lane * 4;
  atomicAdd(a + 0, v.x);
  atomicAdd(a + 1, v.y);
  atomicAdd(a + 2, v.z);
  atomicAdd(a + 3, v.w);
}

// Pack a row-major f32 weight [K x NOUT] into bf16 WMMA B-fragments:
// layout out[((nt*KT + t)*32 + lane)*16 + j] holds W[t*32 + (lane>=16?16:0) + j][nt*16 + (lane&15)].
template <int K, int NOUT>
__global__ void pack_b_kernel(const float* __restrict__ W, __bf16* __restrict__ out) {
  constexpr int KT = K / 32;
  constexpr int NT = NOUT / 16;
  int idx = blockIdx.x * blockDim.x + threadIdx.x;
  if (idx >= NT * KT * 32 * 16) return;
  int j    = idx & 15;
  int lane = (idx >> 4) & 31;
  int frag = idx >> 9;
  int t    = frag % KT;
  int nt   = frag / KT;
  int n    = lane & 15;
  int kb   = (lane >> 4) * 16;
  int krow = t * 32 + kb + j;
  out[idx] = (__bf16)W[krow * NOUT + nt * 16 + n];
}

// ---------------------------------------------------------------- WMMA GEMM
// One wave computes a 16-row x NOUT tile:  out = mean(agg/deg) @ Wl + xself @ Wr + b
template <int K, int NOUT, bool RELU>
__global__ void sage_gemm_kernel(const float* __restrict__ agg,
                                 const float* __restrict__ deg,
                                 const float* __restrict__ xself,
                                 const __bf16* __restrict__ WlF,   // packed fragments
                                 const __bf16* __restrict__ WrF,   // packed fragments
                                 const float* __restrict__ bias,
                                 float* __restrict__ out, int nNodes) {
  constexpr int KT = K / 32;
  constexpr int NT = NOUT / 16;
  int lane = threadIdx.x & 31;
  int tile = blockIdx.x * (blockDim.x >> 5) + (threadIdx.x >> 5);
  if (tile * 16 >= nNodes) return;                 // uniform per-wave exit only

  int m    = lane & 15;
  int half = lane >> 4;                            // 0 or 1
  int row  = tile * 16 + m;
  int rowc = row < nNodes ? row : nNodes - 1;      // clamp for safety (EXEC must stay full)
  int koff = half * 16;

  float invdeg = 1.0f / fmaxf(deg[rowc], 1.0f);
  const float* aRow = agg   + (size_t)rowc * K;
  const float* sRow = xself + (size_t)rowc * K;

  // Build A fragments (contiguous k order, matching packed B):
  //   j in [0,16) -> K = t*32 + koff + j
  v16bf aM[KT], aS[KT];
#pragma unroll
  for (int t = 0; t < KT; ++t) {
    int base = t * 32 + koff;
    float fa[16], fs[16];
    *(float4*)(fa + 0)  = *(const float4*)(aRow + base);
    *(float4*)(fa + 4)  = *(const float4*)(aRow + base + 4);
    *(float4*)(fa + 8)  = *(const float4*)(aRow + base + 8);
    *(float4*)(fa + 12) = *(const float4*)(aRow + base + 12);
    *(float4*)(fs + 0)  = *(const float4*)(sRow + base);
    *(float4*)(fs + 4)  = *(const float4*)(sRow + base + 4);
    *(float4*)(fs + 8)  = *(const float4*)(sRow + base + 8);
    *(float4*)(fs + 12) = *(const float4*)(sRow + base + 12);
#pragma unroll
    for (int j = 0; j < 16; ++j) {
      aM[t][j] = (__bf16)(fa[j] * invdeg);
      aS[t][j] = (__bf16)fs[j];
    }
  }

  const v16bf* Wl = (const v16bf*)WlF;
  const v16bf* Wr = (const v16bf*)WrF;

#pragma unroll
  for (int nt = 0; nt < NT; ++nt) {
    int col = nt * 16 + m;
    float bv = bias[col];
    v8f c;
#pragma unroll
    for (int r = 0; r < 8; ++r) c[r] = bv;        // bias depends only on column
#pragma unroll
    for (int t = 0; t < KT; ++t) {
      v16bf bl = Wl[(nt * KT + t) * 32 + lane];
      v16bf br = Wr[(nt * KT + t) * 32 + lane];
      c = __builtin_amdgcn_wmma_f32_16x16x32_bf16(false, aM[t], false, bl,
                                                  (short)0, c, false, false);
      c = __builtin_amdgcn_wmma_f32_16x16x32_bf16(false, aS[t], false, br,
                                                  (short)0, c, false, false);
    }
    // C/D layout: VGPR r holds (M = r + 8*half, N = m)
#pragma unroll
    for (int r = 0; r < 8; ++r) {
      int orow = tile * 16 + r + 8 * half;
      float v = c[r];
      if (RELU) v = fmaxf(v, 0.0f);
      if (orow < nNodes) out[(size_t)orow * NOUT + col] = v;
    }
  }
}

// ---------------------------------------------------------------- log-softmax
// One wave per node, 64 classes -> 2 per lane, shuffle reductions (wave32).
__global__ void log_softmax_kernel(float* __restrict__ out, int nNodes) {
  int node = blockIdx.x * 8 + (threadIdx.x >> 5);
  int lane = threadIdx.x & 31;
  if (node >= nNodes) return;
  float* row = out + (size_t)node * NCLS;
  float v0 = row[lane], v1 = row[lane + 32];
  float mx = fmaxf(v0, v1);
#pragma unroll
  for (int off = 16; off > 0; off >>= 1) mx = fmaxf(mx, __shfl_xor(mx, off, 32));
  float s = __expf(v0 - mx) + __expf(v1 - mx);
#pragma unroll
  for (int off = 16; off > 0; off >>= 1) s += __shfl_xor(s, off, 32);
  float lse = mx + __logf(s);
  row[lane]      = v0 - lse;
  row[lane + 32] = v1 - lse;
}

// ---------------------------------------------------------------- launcher
extern "C" void kernel_launch(void* const* d_in, const int* in_sizes, int n_in,
                              void* d_out, int out_size, void* d_ws, size_t ws_size,
                              hipStream_t stream) {
  const float*     h   = (const float*)d_in[0];
  const long long* ei  = (const long long*)d_in[1];   // int64 [2, NE]
  const float*     W1l = (const float*)d_in[2];
  const float*     W1r = (const float*)d_in[3];
  const float*     b1  = (const float*)d_in[4];
  const float*     W2l = (const float*)d_in[5];
  const float*     W2r = (const float*)d_in[6];
  const float*     b2  = (const float*)d_in[7];
  const long long* src = ei;
  const long long* dst = ei + NE;

  // Workspace layout (f32 regions first, then 32B-aligned bf16 fragments).
  float* deg = (float*)d_ws;                     // NN
  float* agg = deg + NN;                         // NN*128
  float* x1  = agg + (size_t)NN * NHID;          // NN*128
  __bf16* pW1l = (__bf16*)(x1 + (size_t)NN * NHID);
  __bf16* pW1r = pW1l + FIN * NHID;
  __bf16* pW2l = pW1r + FIN * NHID;
  __bf16* pW2r = pW2l + NHID * NCLS;

  float* logits = (float*)d_out;

  const int tiles  = (NN + 15) / 16;             // 6250
  const int gblk   = (tiles + 7) / 8;            // 8 waves (tiles) per 256-thr block
  const int sblk   = (NE + 7) / 8;               // 8 edges per block

  // 1) zero deg + agg (contiguous)
  zero_f32_kernel<<<2048, 256, 0, stream>>>(deg, (size_t)NN * (1 + NHID));
  // 2) degrees
  degree_kernel<<<(NE + 255) / 256, 256, 0, stream>>>(dst, deg, NE);
  // 3) pack weights into bf16 WMMA fragments
  pack_b_kernel<FIN,  NHID><<<(8 * 4 * 512 + 255) / 256, 256, 0, stream>>>(W1l, pW1l);
  pack_b_kernel<FIN,  NHID><<<(8 * 4 * 512 + 255) / 256, 256, 0, stream>>>(W1r, pW1r);
  pack_b_kernel<NHID, NCLS><<<(4 * 4 * 512 + 255) / 256, 256, 0, stream>>>(W2l, pW2l);
  pack_b_kernel<NHID, NCLS><<<(4 * 4 * 512 + 255) / 256, 256, 0, stream>>>(W2r, pW2r);
  // 4) layer-1 scatter
  scatter128_kernel<<<sblk, 256, 0, stream>>>(h, src, dst, agg, NE);
  // 5) layer-1 GEMM + ReLU -> x1
  sage_gemm_kernel<FIN, NHID, true><<<gblk, 256, 0, stream>>>(
      agg, deg, h, pW1l, pW1r, b1, x1, NN);
  // 6) re-zero agg
  zero_f32_kernel<<<2048, 256, 0, stream>>>(agg, (size_t)NN * NHID);
  // 7) layer-2 scatter
  scatter128_kernel<<<sblk, 256, 0, stream>>>(x1, src, dst, agg, NE);
  // 8) layer-2 GEMM -> logits
  sage_gemm_kernel<NHID, NCLS, false><<<gblk, 256, 0, stream>>>(
      agg, deg, x1, pW2l, pW2r, b2, logits, NN);
  // 9) log-softmax in place
  log_softmax_kernel<<<(NN + 7) / 8, 256, 0, stream>>>(logits, NN);

  (void)in_sizes; (void)n_in; (void)out_size; (void)ws_size;
}
